// GATSurvivalModel_65206193488369
// MI455X (gfx1250) — compile-verified, matrix operand order checked by reading, not run on hardware
//
#include <hip/hip_runtime.h>
#include <hip/hip_bf16.h>

typedef __attribute__((ext_vector_type(16))) _Float16 v16h;
typedef __attribute__((ext_vector_type(8)))  _Float16 v8h;
typedef __attribute__((ext_vector_type(8)))  float    v8f;

// ---------------------------------------------------------------- helpers
__device__ __forceinline__ float seluf(float x) {
    const float a = 1.6732632423543772f, s = 1.0507009873554805f;
    return s * (x > 0.f ? x : a * (__expf(x) - 1.f));
}
__device__ __forceinline__ float eluf(float x) { return x > 0.f ? x : __expf(x) - 1.f; }

__device__ __forceinline__ float wave_sum32(float v) {
    #pragma unroll
    for (int m = 16; m > 0; m >>= 1) v += __shfl_xor(v, m, 32);
    return v;
}

// ---------------------------------------------------------------- utility kernels
__global__ void fillz(float* __restrict__ p, int n) {
    int i = blockIdx.x * blockDim.x + threadIdx.x;
    if (i < n) p[i] = 0.f;
}

__global__ void cvt_f16(const float* __restrict__ in, _Float16* __restrict__ out, int n) {
    int i = blockIdx.x * blockDim.x + threadIdx.x;
    if (i < n) out[i] = (_Float16)in[i];
}

// ---------------------------------------------------------------- WMMA GEMM
// C[r][c] = act( sum_k A[r][k]*B[c][k] + bias[c] )
// A: nrows x K (f16 row-major), B: Cout x K (f16 row-major, i.e. W as given)
// grid.x = ceil(nrows/16); block = 32 * (Cout/16): one wave per 16x16 output tile.
// Fragment layout (ISA 7.12.2, 16-bit A/B): per lane the 16 halves are two
// contiguous runs of 8 K-values starting at 8*hi and 16+8*hi -> two b128 loads.
// act: 0 = none, 1 = SELU.  Ch (optional): f16 copy of C for chained GEMMs.
__global__ void wmma_gemm_f16(const _Float16* __restrict__ A,
                              const _Float16* __restrict__ B,
                              const float* __restrict__ bias,
                              float* __restrict__ C,
                              _Float16* __restrict__ Ch,
                              int nrows, int K, int Cout, int act)
{
    const int lane = threadIdx.x & 31;
    const int wave = threadIdx.x >> 5;
    const int mn   = lane & 15;
    const int hi   = lane >> 4;
    const int row0 = blockIdx.x * 16;
    const int col0 = wave * 16;

    int ar = row0 + mn;
    if (ar >= nrows) ar = nrows - 1;          // clamp: loads always valid, stores guarded
    const _Float16* Ap = A + (size_t)ar * K + 8 * hi;
    const _Float16* Bp = B + (size_t)(col0 + mn) * K + 8 * hi;

    v8f acc = {};
    for (int kb = 0; kb < K; kb += 32) {
        v8h a0 = *(const v8h*)(Ap + kb);
        v8h a1 = *(const v8h*)(Ap + kb + 16);
        v8h b0 = *(const v8h*)(Bp + kb);
        v8h b1 = *(const v8h*)(Bp + kb + 16);
        v16h a = __builtin_shufflevector(a0, a1, 0,1,2,3,4,5,6,7,8,9,10,11,12,13,14,15);
        v16h b = __builtin_shufflevector(b0, b1, 0,1,2,3,4,5,6,7,8,9,10,11,12,13,14,15);
        acc = __builtin_amdgcn_wmma_f32_16x16x32_f16(
            /*neg_a=*/false, a, /*neg_b=*/false, b,
            /*c_mod=*/(short)0, acc, /*reuse_a=*/false, /*reuse_b=*/false);
    }

    // C/D layout: VGPR v -> M = v + 8*hi ; N = lane & 15
    const int c = col0 + mn;
    #pragma unroll
    for (int v = 0; v < 8; ++v) {
        int r = row0 + v + 8 * hi;
        if (r < nrows) {
            float val = acc[v];
            if (bias) val += bias[c];
            if (act == 1) val = seluf(val);
            C[(size_t)r * Cout + c] = val;
            if (Ch) Ch[(size_t)r * Cout + c] = (_Float16)val;
        }
    }
}

// ---------------------------------------------------------------- attention scores
// H: n x (NH*D) f32 ; a_src/a_dst: NH x D.  One thread per (node, head).
__global__ void attn_scores(const float* __restrict__ H,
                            const float* __restrict__ a_src,
                            const float* __restrict__ a_dst,
                            float* __restrict__ es, float* __restrict__ ed,
                            int n, int NHv, int D)
{
    int idx = blockIdx.x * blockDim.x + threadIdx.x;
    if (idx >= n * NHv) return;
    int node = idx / NHv, h = idx - node * NHv;
    const float* hp = H + (size_t)node * NHv * D + (size_t)h * D;
    const float* as = a_src + h * D;
    const float* ad = a_dst + h * D;
    float s = 0.f, d = 0.f;
    for (int i = 0; i < D; ++i) { float v = hp[i]; s += v * as[i]; d += v * ad[i]; }
    es[idx] = s; ed[idx] = d;
}

// ---------------------------------------------------------------- edge pass 1
// e = leaky_relu(es[src]+ed[dst]) * (1+et_emb[et]) * ew ; atomic max (clamped at 0).
// emax starts at +0.0f; only positive values can change max(seg_max, 0), and for
// non-negative IEEE floats uint bit-order == float order -> atomicMax on u32.
__global__ void edge_logits(const int* __restrict__ src, const int* __restrict__ dst,
                            const int* __restrict__ et, const float* __restrict__ ew,
                            const float* __restrict__ esv, const float* __restrict__ edv,
                            const float* __restrict__ etEmb,
                            float* __restrict__ eE, unsigned* __restrict__ emax,
                            int E_, int NHv)
{
    int e = blockIdx.x * blockDim.x + threadIdx.x;
    if (e >= E_) return;
    int s = src[e], d = dst[e], t = et[e];
    float w = ew[e];
    for (int h = 0; h < NHv; ++h) {
        float v = esv[s * NHv + h] + edv[d * NHv + h];
        v = (v > 0.f) ? v : 0.2f * v;                 // leaky_relu(0.2)
        v *= (1.f + etEmb[t * NHv + h]);
        v *= w;
        eE[(size_t)e * NHv + h] = v;
        if (v > 0.f) atomicMax(emax + (size_t)d * NHv + h, __float_as_uint(v));
    }
}

// ---------------------------------------------------------------- edge pass 2
__global__ void edge_expsum(const int* __restrict__ dst, const float* __restrict__ eE,
                            const unsigned* __restrict__ emax, float* __restrict__ esum,
                            int E_, int NHv)
{
    int e = blockIdx.x * blockDim.x + threadIdx.x;
    if (e >= E_) return;
    int d = dst[e];
    for (int h = 0; h < NHv; ++h) {
        float m = __uint_as_float(emax[(size_t)d * NHv + h]);
        atomicAdd(esum + (size_t)d * NHv + h, __expf(eE[(size_t)e * NHv + h] - m));
    }
}

// ---------------------------------------------------------------- edge pass 3
// One wave32 per edge: lane handles 4 consecutive features (F = NH*D = 128 always).
__global__ void edge_aggregate(const int* __restrict__ src, const int* __restrict__ dst,
                               const float* __restrict__ eE,
                               const unsigned* __restrict__ emax,
                               const float* __restrict__ esum,
                               const float* __restrict__ H, float* __restrict__ agg,
                               int E_, int NHv, int D)
{
    int lane = threadIdx.x & 31;
    int e = blockIdx.x * (blockDim.x >> 5) + (threadIdx.x >> 5);
    if (e >= E_) return;
    int s = src[e], d = dst[e];
    int f = lane * 4;
    int h = f / D;                                    // D in {32,128}; quad stays in one head
    float m  = __uint_as_float(emax[(size_t)d * NHv + h]);
    float sm = esum[(size_t)d * NHv + h];
    float alpha = __expf(eE[(size_t)e * NHv + h] - m) / (sm + 1e-10f);
    const float4 hv = *(const float4*)(H + (size_t)s * 128 + f);
    float* o = agg + (size_t)d * 128 + f;
    atomicAdd(o + 0, hv.x * alpha);
    atomicAdd(o + 1, hv.y * alpha);
    atomicAdd(o + 2, hv.z * alpha);
    atomicAdd(o + 3, hv.w * alpha);
}

// ---------------------------------------------------------------- residual + LayerNorm (+ELU)
// One wave32 per node (32 lanes x float4 = 128 features). Writes f32 + f16 copies.
__global__ void ln_residual(const float* __restrict__ agg, const float* __restrict__ xres,
                            const float* __restrict__ g, const float* __restrict__ b,
                            float* __restrict__ outF, _Float16* __restrict__ outH,
                            int n, int doElu)
{
    int lane = threadIdx.x & 31;
    int node = blockIdx.x * (blockDim.x >> 5) + (threadIdx.x >> 5);
    if (node >= n) return;
    int f = lane * 4;
    const float4 av = *(const float4*)(agg  + (size_t)node * 128 + f);
    const float4 xv = *(const float4*)(xres + (size_t)node * 128 + f);
    float4 v = make_float4(av.x + xv.x, av.y + xv.y, av.z + xv.z, av.w + xv.w);

    float mean = wave_sum32(v.x + v.y + v.z + v.w) * (1.f / 128.f);
    float4 c = make_float4(v.x - mean, v.y - mean, v.z - mean, v.w - mean);
    float var = wave_sum32(c.x * c.x + c.y * c.y + c.z * c.z + c.w * c.w) * (1.f / 128.f);
    float rstd = rsqrtf(var + 1e-5f);

    const float4 gg = *(const float4*)(g + f);
    const float4 bb = *(const float4*)(b + f);
    float4 y = make_float4(c.x * rstd * gg.x + bb.x, c.y * rstd * gg.y + bb.y,
                           c.z * rstd * gg.z + bb.z, c.w * rstd * gg.w + bb.w);
    if (doElu) { y.x = eluf(y.x); y.y = eluf(y.y); y.z = eluf(y.z); y.w = eluf(y.w); }

    *(float4*)(outF + (size_t)node * 128 + f) = y;
    _Float16* oh = outH + (size_t)node * 128 + f;
    oh[0] = (_Float16)y.x; oh[1] = (_Float16)y.y;
    oh[2] = (_Float16)y.z; oh[3] = (_Float16)y.w;
}

// ---------------------------------------------------------------- survival head tail
// One thread per node; rw2 / (rb2+baseline) staged in LDS; running recurrences for
// cum_hazard, survival, shifted, and both CIF channels.
__global__ void surv_tail(const float* __restrict__ lhs, const float* __restrict__ rw2,
                          const float* __restrict__ rb2, const float* __restrict__ baseline,
                          float* __restrict__ haz, float* __restrict__ surv,
                          float* __restrict__ cif, float* __restrict__ es,
                          float* __restrict__ er, int n)
{
    __shared__ float s_w[2 * 60 * 16];
    __shared__ float s_b[2 * 60];
    for (int i = threadIdx.x; i < 2 * 60 * 16; i += blockDim.x) s_w[i] = rw2[i];
    for (int i = threadIdx.x; i < 2 * 60; i += blockDim.x) s_b[i] = rb2[i] + baseline[i];
    __syncthreads();

    int node = blockIdx.x * blockDim.x + threadIdx.x;
    if (node >= n) return;

    float l0[16], l1[16];
    const float* lp = lhs + (size_t)node * 32;
    #pragma unroll
    for (int k = 0; k < 16; ++k) { l0[k] = lp[k]; l1[k] = lp[16 + k]; }

    float cum = 0.f, shifted = 1.f, c0 = 0.f, c1 = 0.f;
    for (int t = 0; t < 60; ++t) {
        float z0 = s_b[t], z1 = s_b[60 + t];
        const float* w0 = s_w + t * 16;            // rw2[0][t][:]
        const float* w1 = s_w + 960 + t * 16;      // rw2[1][t][:]
        #pragma unroll
        for (int k = 0; k < 16; ++k) { z0 += l0[k] * w0[k]; z1 += l1[k] * w1[k]; }
        float h0 = (z0 > 20.f) ? z0 : log1pf(__expf(z0));   // softplus
        float h1 = (z1 > 20.f) ? z1 : log1pf(__expf(z1));
        cum += h0 + h1;
        float sv = __expf(-cum);
        c0 += h0 * shifted; c1 += h1 * shifted;
        haz [(size_t)node * 120 + t]      = h0;
        haz [(size_t)node * 120 + 60 + t] = h1;
        surv[(size_t)node * 60  + t]      = sv;
        cif [(size_t)node * 120 + t]      = c0;
        cif [(size_t)node * 120 + 60 + t] = c1;
        shifted = sv;
    }
    es[node] = c0 / (c0 + c1 + 1e-10f);
    er[node] = c0 * 0.1f + c1 * (-0.05f);
}

// ---------------------------------------------------------------- launcher
extern "C" void kernel_launch(void* const* d_in, const int* in_sizes, int n_in,
                              void* d_out, int out_size, void* d_ws, size_t ws_size,
                              hipStream_t stream)
{
    const int n  = in_sizes[0] / 128;   // 20000
    const int E_ = in_sizes[2];         // 640000

    const float* x    = (const float*)d_in[0];
    const int* eidx   = (const int*)d_in[1];
    const int* src_i  = eidx;
    const int* dst_i  = eidx + E_;
    const int* etyp   = (const int*)d_in[2];
    const float* ew   = (const float*)d_in[3];

    const float *LW[3], *LAS[3], *LAD[3], *LEE[3], *LG[3], *LB[3];
    for (int i = 0; i < 3; ++i) {
        int b = 4 + i * 6;
        LW[i]  = (const float*)d_in[b];
        LAS[i] = (const float*)d_in[b + 1];
        LAD[i] = (const float*)d_in[b + 2];
        LEE[i] = (const float*)d_in[b + 3];
        LG[i]  = (const float*)d_in[b + 4];
        LB[i]  = (const float*)d_in[b + 5];
    }
    const float* w1  = (const float*)d_in[22]; const float* b1  = (const float*)d_in[23];
    const float* w2  = (const float*)d_in[24]; const float* b2  = (const float*)d_in[25];
    const float* rw1 = (const float*)d_in[26]; const float* rb1 = (const float*)d_in[27];
    const float* rw2 = (const float*)d_in[28]; const float* rb2 = (const float*)d_in[29];
    const float* base= (const float*)d_in[30];

    // outputs (tuple order: emb, hazards, survival, cif, entry_score, expected_return)
    float* o_emb  = (float*)d_out;
    float* o_haz  = o_emb  + (size_t)n * 128;
    float* o_surv = o_haz  + (size_t)n * 120;
    float* o_cif  = o_surv + (size_t)n * 60;
    float* o_es   = o_cif  + (size_t)n * 120;
    float* o_er   = o_es   + n;

    // workspace carve
    char* wp = (char*)d_ws;
    auto carve = [&](size_t bytes) -> void* {
        void* r = (void*)wp; wp += (bytes + 255) & ~size_t(255); return r;
    };
    float*    H     = (float*)carve((size_t)n * 128 * 4);
    float*    agg   = (float*)carve((size_t)n * 128 * 4);
    float*    nodeA = (float*)carve((size_t)n * 128 * 4);
    float*    nodeB = (float*)carve((size_t)n * 128 * 4);
    _Float16* xh    = (_Float16*)carve((size_t)n * 128 * 2);
    _Float16* wh    = (_Float16*)carve(128 * 128 * 2);
    float*    eSrc  = (float*)carve((size_t)n * 4 * 4);
    float*    eDst  = (float*)carve((size_t)n * 4 * 4);
    unsigned* eMax  = (unsigned*)carve((size_t)n * 4 * 4);
    float*    eSum  = (float*)carve((size_t)n * 4 * 4);
    float*    eE    = (float*)carve((size_t)E_ * 4 * 4);
    float*    s1    = (float*)carve((size_t)n * 32 * 4);
    float*    s2    = (float*)carve((size_t)n * 32 * 4);
    float*    lhs   = (float*)carve((size_t)n * 32 * 4);
    _Float16* s1h   = (_Float16*)carve((size_t)n * 32 * 2);
    _Float16* s2h   = (_Float16*)carve((size_t)n * 32 * 2);

    const int TB = 256;
    cvt_f16<<<(n * 128 + TB - 1) / TB, TB, 0, stream>>>(x, xh, n * 128);

    for (int layer = 0; layer < 3; ++layer) {
        const int NHv = (layer < 2) ? 4 : 1;
        const int D   = (layer < 2) ? 32 : 128;
        const float* xres = (layer == 0) ? x : (layer == 1 ? nodeA : nodeB);

        cvt_f16<<<(128 * 128 + TB - 1) / TB, TB, 0, stream>>>(LW[layer], wh, 128 * 128);
        // one wave per 16x16 tile; 8 col-tiles (Cout=128) -> 8 waves per block
        wmma_gemm_f16<<<(n + 15) / 16, 32 * 8, 0, stream>>>(
            xh, wh, nullptr, H, nullptr, n, 128, 128, 0);

        attn_scores<<<(n * NHv + TB - 1) / TB, TB, 0, stream>>>(
            H, LAS[layer], LAD[layer], eSrc, eDst, n, NHv, D);

        fillz<<<(n * NHv + TB - 1) / TB, TB, 0, stream>>>((float*)eMax, n * NHv);
        fillz<<<(n * NHv + TB - 1) / TB, TB, 0, stream>>>(eSum, n * NHv);
        fillz<<<(n * 128 + TB - 1) / TB, TB, 0, stream>>>(agg, n * 128);

        edge_logits<<<(E_ + TB - 1) / TB, TB, 0, stream>>>(
            src_i, dst_i, etyp, ew, eSrc, eDst, LEE[layer], eE, eMax, E_, NHv);
        edge_expsum<<<(E_ + TB - 1) / TB, TB, 0, stream>>>(dst_i, eE, eMax, eSum, E_, NHv);
        edge_aggregate<<<(E_ + 7) / 8, 256, 0, stream>>>(
            src_i, dst_i, eE, eMax, eSum, H, agg, E_, NHv, D);

        float* outF = (layer == 0) ? nodeA : (layer == 1 ? nodeB : o_emb);
        ln_residual<<<(n + 7) / 8, 256, 0, stream>>>(
            agg, xres, LG[layer], LB[layer], outF, xh, n, (layer < 2) ? 1 : 0);
    }

    // ---- survival head ----
    cvt_f16<<<(32 * 128 + TB - 1) / TB, TB, 0, stream>>>(w1, wh, 32 * 128);
    wmma_gemm_f16<<<(n + 15) / 16, 32 * 2, 0, stream>>>(xh, wh, b1, s1, s1h, n, 128, 32, 1);

    cvt_f16<<<(32 * 32 + TB - 1) / TB, TB, 0, stream>>>(w2, wh, 32 * 32);
    wmma_gemm_f16<<<(n + 15) / 16, 32 * 2, 0, stream>>>(s1h, wh, b2, s2, s2h, n, 32, 32, 1);

    cvt_f16<<<(32 * 32 + TB - 1) / TB, TB, 0, stream>>>(rw1, wh, 32 * 32);
    wmma_gemm_f16<<<(n + 15) / 16, 32 * 2, 0, stream>>>(s2h, wh, rb1, lhs, nullptr, n, 32, 32, 1);

    surv_tail<<<(n + 127) / 128, 128, 0, stream>>>(
        lhs, rw2, rb2, base, o_haz, o_surv, o_cif, o_es, o_er, n);
}
